// RNDiscriminator_69930657513578
// MI455X (gfx1250) — compile-verified, matrix operand order checked by reading, not run on hardware
//
#include <hip/hip_runtime.h>
#include <hip/hip_bf16.h>

typedef __attribute__((ext_vector_type(16))) __bf16 v16bf;
typedef __attribute__((ext_vector_type(8)))  float  v8f;

union ABFrag {
    v16bf    bf;
    unsigned u[8];
};

__device__ __forceinline__ unsigned short f32_to_bf16_rne(float f) {
    union { float f; unsigned u; } x;
    x.f = f;
    unsigned r = x.u + 0x7FFFu + ((x.u >> 16) & 1u);   // round-to-nearest-even
    return (unsigned short)(r >> 16);
}

// ---------------------------------------------------------------------------
// Kernel 1: conv(8x8,s8)+ReLU+coords for one (batch, position) per block,
// then project the 26-vector through W1's top/bottom halves -> u, v.
// ---------------------------------------------------------------------------
__global__ __launch_bounds__(256) void rn_features(
    const float* __restrict__ image,   // (64,3,64,64)
    const float* __restrict__ conv_w,  // (24,3,8,8)
    const float* __restrict__ conv_b,  // (24,)
    const float* __restrict__ g_w1,    // (52,256)
    float* __restrict__ u,             // (64,64,256)
    float* __restrict__ v)             // (64,64,256)
{
    __shared__ float xf[26];
    const int b  = blockIdx.x;     // batch
    const int p  = blockIdx.y;     // position 0..63  (p = py*8 + px)
    const int py = p >> 3, px = p & 7;
    const int t  = threadIdx.x;

    if (t < 24) {
        const float* w = conv_w + t * 192;
        float acc = conv_b[t];
        #pragma unroll
        for (int ci = 0; ci < 3; ++ci) {
            const float* img = image + ((size_t)b * 3 + ci) * 4096 + py * 8 * 64 + px * 8;
            const float* wk  = w + ci * 64;
            #pragma unroll
            for (int ky = 0; ky < 8; ++ky)
                #pragma unroll
                for (int kx = 0; kx < 8; ++kx)
                    acc += img[ky * 64 + kx] * wk[ky * 8 + kx];
        }
        xf[t] = acc > 0.f ? acc : 0.f;
    } else if (t == 24) {
        xf[24] = -1.0f + (2.0f / 7.0f) * (float)px;   // cx: varies along columns
    } else if (t == 25) {
        xf[25] = -1.0f + (2.0f / 7.0f) * (float)py;   // cy: varies along rows
    }
    __syncthreads();

    // thread t = output channel n (0..255)
    float su = 0.f, sv = 0.f;
    #pragma unroll
    for (int c = 0; c < 26; ++c) {
        const float x = xf[c];
        su += x * g_w1[c * 256 + t];            // W1 rows 0..25
        sv += x * g_w1[(c + 26) * 256 + t];     // W1 rows 26..51
    }
    const size_t off = ((size_t)b * 64 + p) * 256 + t;
    u[off] = su;
    v[off] = sv;
}

// ---------------------------------------------------------------------------
// Kernel 2: W2 (256x256 f32, [k][n]) -> bf16 transposed W2t[n][k]; zero pooled.
// ---------------------------------------------------------------------------
__global__ __launch_bounds__(256) void rn_prep(
    const float* __restrict__ g_w2,
    unsigned short* __restrict__ W2t,
    float* __restrict__ pooled)
{
    const int idx = blockIdx.x * 256 + threadIdx.x;   // 0..65535
    const int k = idx >> 8, n = idx & 255;
    W2t[n * 256 + k] = f32_to_bf16_rne(g_w2[k * 256 + n]);
    if (idx < 64 * 256) pooled[idx] = 0.f;
}

// ---------------------------------------------------------------------------
// Kernel 3: the dominant GEMM. One block per (batch b, left index i).
// h1[j][:] = relu(u[b,j] + v[b,i] + b1) built as bf16 in LDS (64x256),
// then h2 = relu(h1 @ W2 + b2) summed over rows via WMMA and atomics.
// ---------------------------------------------------------------------------
#define H1_RS 264   // padded row stride (elements) to avoid LDS bank conflicts

__global__ __launch_bounds__(256) void rn_gemm(
    const float* __restrict__ u,
    const float* __restrict__ v,
    const float* __restrict__ g_b1,
    const unsigned short* __restrict__ W2t,   // bf16 [n][k]
    const float* __restrict__ g_b2,
    float* __restrict__ pooled)               // (64,256) f32
{
    __shared__ unsigned short h1[64 * H1_RS];

    const int b = blockIdx.x;    // batch
    const int i = blockIdx.y;    // left pair index
    const int t = threadIdx.x;

    // ---- build h1 tile: thread t handles row j=t/4, cols (t%4)*64..+63 ----
    {
        const int j  = t >> 2;
        const int c0 = (t & 3) * 64;
        const float* up = u + (((size_t)b * 64 + j) * 256) + c0;
        const float* vp = v + (((size_t)b * 64 + i) * 256) + c0;
        const float* bp = g_b1 + c0;
        #pragma unroll 8
        for (int c = 0; c < 64; ++c) {
            float x = up[c] + vp[c] + bp[c];
            x = x > 0.f ? x : 0.f;
            h1[j * H1_RS + c0 + c] = f32_to_bf16_rne(x);
        }
    }
    __syncthreads();

    const int wave = t >> 5;          // 0..7, owns N slice [wave*32, wave*32+32)
    const int lane = t & 31;
    const int half = lane >> 4;       // 0: lanes 0-15, 1: lanes 16-31
    const int ln   = lane & 15;
    const int n0   = wave * 32;

    v8f acc[4][2] = {};

    for (int kc = 0; kc < 8; ++kc) {                  // K chunks of 32
        ABFrag bf0, bf1;
        #pragma unroll
        for (int vv = 0; vv < 8; ++vv) {
            const int k0 = kc * 32 + ((vv < 4) ? 2 * vv : 16 + 2 * (vv - 4)) + half * 8;
            bf0.u[vv] = *(const unsigned*)(W2t + (size_t)(n0 + ln) * 256 + k0);
            bf1.u[vv] = *(const unsigned*)(W2t + (size_t)(n0 + 16 + ln) * 256 + k0);
        }
        #pragma unroll
        for (int mt = 0; mt < 4; ++mt) {              // 4 M-tiles of 16 rows
            ABFrag af;
            #pragma unroll
            for (int vv = 0; vv < 8; ++vv) {
                const int k0 = kc * 32 + ((vv < 4) ? 2 * vv : 16 + 2 * (vv - 4)) + half * 8;
                af.u[vv] = *(const unsigned*)&h1[(mt * 16 + ln) * H1_RS + k0];
            }
            acc[mt][0] = __builtin_amdgcn_wmma_f32_16x16x32_bf16(
                false, af.bf, false, bf0.bf, (short)0, acc[mt][0], false, false);
            acc[mt][1] = __builtin_amdgcn_wmma_f32_16x16x32_bf16(
                false, af.bf, false, bf1.bf, (short)0, acc[mt][1], false, false);
        }
    }

    // ---- bias + ReLU + column sums over the 64 rows this block owns ----
    float s0 = 0.f, s1 = 0.f;
    const float bias0 = g_b2[n0 + ln];
    const float bias1 = g_b2[n0 + 16 + ln];
    #pragma unroll
    for (int mt = 0; mt < 4; ++mt) {
        #pragma unroll
        for (int r = 0; r < 8; ++r) {
            const float x0 = acc[mt][0][r] + bias0; s0 += x0 > 0.f ? x0 : 0.f;
            const float x1 = acc[mt][1][r] + bias1; s1 += x1 > 0.f ? x1 : 0.f;
        }
    }
    // lanes l and l+16 hold the same n (different m halves): fold halves
    s0 += __shfl_xor(s0, 16, 32);
    s1 += __shfl_xor(s1, 16, 32);
    if (half == 0) {
        atomicAdd(&pooled[b * 256 + n0 + ln],       s0);
        atomicAdd(&pooled[b * 256 + n0 + 16 + ln],  s1);
    }
}

// ---------------------------------------------------------------------------
// Kernel 4: f head, one block per batch. out[b] = relu(pooled@f_w1+b)@f_w2+b
// ---------------------------------------------------------------------------
__global__ __launch_bounds__(256) void rn_head(
    const float* __restrict__ pooled,
    const float* __restrict__ f_w1,   // (256,256)
    const float* __restrict__ f_b1,
    const float* __restrict__ f_w2,   // (256,1)
    const float* __restrict__ f_b2,
    float* __restrict__ out)          // (64,1)
{
    __shared__ float red[256];
    const int b = blockIdx.x, n = threadIdx.x;
    const float* pb = pooled + b * 256;
    float h = f_b1[n];
    for (int k = 0; k < 256; ++k) h += pb[k] * f_w1[k * 256 + n];
    h = h > 0.f ? h : 0.f;
    red[n] = h * f_w2[n];
    __syncthreads();
    for (int s = 128; s > 0; s >>= 1) {
        if (n < s) red[n] += red[n + s];
        __syncthreads();
    }
    if (n == 0) out[b] = red[0] + f_b2[0];
}

// ---------------------------------------------------------------------------
extern "C" void kernel_launch(void* const* d_in, const int* in_sizes, int n_in,
                              void* d_out, int out_size, void* d_ws, size_t ws_size,
                              hipStream_t stream) {
    (void)in_sizes; (void)n_in; (void)out_size; (void)ws_size;
    const float* image  = (const float*)d_in[0];
    const float* conv_w = (const float*)d_in[1];
    const float* conv_b = (const float*)d_in[2];
    const float* g_w1   = (const float*)d_in[3];
    const float* g_b1   = (const float*)d_in[4];
    const float* g_w2   = (const float*)d_in[5];
    const float* g_b2   = (const float*)d_in[6];
    const float* f_w1   = (const float*)d_in[7];
    const float* f_b1   = (const float*)d_in[8];
    const float* f_w2   = (const float*)d_in[9];
    const float* f_b2   = (const float*)d_in[10];
    float* out = (float*)d_out;

    char* ws = (char*)d_ws;
    float*          u      = (float*)(ws);                            // 4 MB
    float*          v      = (float*)(ws + (size_t)4 * 1024 * 1024);  // 4 MB
    float*          pooled = (float*)(ws + (size_t)8 * 1024 * 1024);  // 64 KB
    unsigned short* W2t    = (unsigned short*)(ws + (size_t)8 * 1024 * 1024 + 64 * 1024); // 128 KB

    rn_features<<<dim3(64, 64), 256, 0, stream>>>(image, conv_w, conv_b, g_w1, u, v);
    rn_prep<<<256, 256, 0, stream>>>(g_w2, W2t, pooled);
    rn_gemm<<<dim3(64, 64), 256, 0, stream>>>(u, v, g_b1, W2t, g_b2, pooled);
    rn_head<<<64, 256, 0, stream>>>(pooled, f_w1, f_b1, f_w2, f_b2, out);
}
